// Transformer_Q_56693568307508
// MI455X (gfx1250) — matrix operand order repaired and checked by reading, not compiled, not run on hardware
//
#include <hip/hip_runtime.h>

// ---------------------------------------------------------------------------
// DETR-style transformer forward for MI455X (gfx1250), all matmuls via
// v_wmma_f32_16x16x32_f16 (wave32 WMMA, fp32 accumulate, fp32 I/O with
// convert-on-load). Encoder attention is fused in-LDS (32x1024 fp32 score
// tile = 128KB, legal only on CDNA5's 320KB LDS). Cross "assignment"
// attention materializes scores for softmax-over-q and head-mean outputs.
// ---------------------------------------------------------------------------

typedef __attribute__((ext_vector_type(16))) _Float16 v16h;
typedef __attribute__((ext_vector_type(8)))  float    v8f;

#define D_MODEL 512
#define DFF_    2048
#define NHEADS  8
#define DH      64
#define SEQLEN  1024
#define BATCHB  8
#define NQUERY  256

#define V8F_ZERO {0.f,0.f,0.f,0.f,0.f,0.f,0.f,0.f}

__device__ __forceinline__ v8f wmma16(v16h a, v16h b, v8f c) {
  return __builtin_amdgcn_wmma_f32_16x16x32_f16(false, a, false, b, (short)0, c, false, false);
}

// A fragment: 16x32 f16, rows m0..m0+15, K window [k0,k0+32).
// lane = grp*16 + r : row = m0+r ; element t -> K = k0 + grp*16 + t
__device__ __forceinline__ v16h load_a_frag_g(const float* __restrict__ base, long lda,
                                              int m0, int k0, int lane) {
  int r = lane & 15, grp = lane >> 4;
  const float* p = base + (long)(m0 + r) * lda + k0 + grp * 16;
  float t[16];
  *(float4*)(t + 0)  = *(const float4*)(p + 0);
  *(float4*)(t + 4)  = *(const float4*)(p + 4);
  *(float4*)(t + 8)  = *(const float4*)(p + 8);
  *(float4*)(t + 12) = *(const float4*)(p + 12);
  v16h a;
#pragma unroll
  for (int i = 0; i < 16; ++i) a[i] = (_Float16)t[i];
  return a;
}

// B fragment (W stored [N,K] row-major, i.e. B^T GEMM):
// lane = grp*16 + c : column n0+c ; element t -> K = k0 + grp*16 + t
__device__ __forceinline__ v16h load_bt_frag_g(const float* __restrict__ base, long ldb,
                                               int n0, int k0, int lane) {
  int c = lane & 15, grp = lane >> 4;
  const float* p = base + (long)(n0 + c) * ldb + k0 + grp * 16;
  float t[16];
  *(float4*)(t + 0)  = *(const float4*)(p + 0);
  *(float4*)(t + 4)  = *(const float4*)(p + 4);
  *(float4*)(t + 8)  = *(const float4*)(p + 8);
  *(float4*)(t + 12) = *(const float4*)(p + 12);
  v16h b;
#pragma unroll
  for (int i = 0; i < 16; ++i) b[i] = (_Float16)t[i];
  return b;
}

// B fragment (V stored [K,N] row-major, non-transposed): strided scalar loads.
__device__ __forceinline__ v16h load_bn_frag_g(const float* __restrict__ base, long ldb,
                                               int n0, int k0, int lane) {
  int c = lane & 15, grp = lane >> 4;
  const float* p = base + (long)(k0 + grp * 16) * ldb + n0 + c;
  v16h b;
#pragma unroll
  for (int t = 0; t < 16; ++t) b[t] = (_Float16)p[(long)t * ldb];
  return b;
}

// A fragment from LDS fp32 tile (attention probabilities).
__device__ __forceinline__ v16h load_a_frag_lds(const float* s, int lds,
                                                int m0, int k0, int lane) {
  int r = lane & 15, grp = lane >> 4;
  const float* p = s + (long)(m0 + r) * lds + k0 + grp * 16;
  v16h a;
#pragma unroll
  for (int i = 0; i < 8; ++i) a[i] = (_Float16)p[i];
#pragma unroll
  for (int i = 0; i < 8; ++i) a[8 + i] = (_Float16)p[8 + i];
  return a;
}

// ---------------------------------------------------------------------------
// Batched WMMA GEMM: C[m,n] = scale * sum_k A[m,k] * B(n,k) + bias[n]
//   BT=true : B stored [N,K] row-major (weights, K^T)
//   BT=false: B stored [K,N] row-major (P @ V)
// Block = 128 threads (4 waves), 64x64 tile, each wave a 32x32 subtile.
// All M,N multiples of 64; K multiple of 32.
// ---------------------------------------------------------------------------
template <bool BT, bool RELU>
__global__ __launch_bounds__(128)
void gemm_wmma(const float* __restrict__ A, long lda, long aB,
               const float* __restrict__ Bm, long ldb, long bB,
               const float* __restrict__ bias,
               float* __restrict__ C, long ldc, long cB,
               int K, float scale) {
  const float* Ab = A + (long)blockIdx.z * aB;
  const float* Bb = Bm + (long)blockIdx.z * bB;
  float* Cb = C + (long)blockIdx.z * cB;
  int lane = threadIdx.x & 31;
  int wave = threadIdx.x >> 5;
  int m0 = blockIdx.y * 64 + (wave >> 1) * 32;
  int n0 = blockIdx.x * 64 + (wave & 1) * 32;
  v8f c00 = V8F_ZERO, c01 = V8F_ZERO, c10 = V8F_ZERO, c11 = V8F_ZERO;
  for (int k0 = 0; k0 < K; k0 += 32) {
    v16h a0 = load_a_frag_g(Ab, lda, m0, k0, lane);
    v16h a1 = load_a_frag_g(Ab, lda, m0 + 16, k0, lane);
    v16h b0, b1;
    if (BT) {
      b0 = load_bt_frag_g(Bb, ldb, n0, k0, lane);
      b1 = load_bt_frag_g(Bb, ldb, n0 + 16, k0, lane);
    } else {
      b0 = load_bn_frag_g(Bb, ldb, n0, k0, lane);
      b1 = load_bn_frag_g(Bb, ldb, n0 + 16, k0, lane);
    }
    c00 = wmma16(a0, b0, c00);
    c01 = wmma16(a0, b1, c01);
    c10 = wmma16(a1, b0, c10);
    c11 = wmma16(a1, b1, c11);
  }
  int grp = lane >> 4, cl = lane & 15;
  int ca0 = n0 + cl, ca1 = n0 + 16 + cl;
  float bb0 = bias ? bias[ca0] : 0.f;
  float bb1 = bias ? bias[ca1] : 0.f;
#pragma unroll
  for (int v = 0; v < 8; ++v) {
    int r0 = m0 + grp * 8 + v;
    int r1 = r0 + 16;
    float x;
    x = c00[v] * scale + bb0; if (RELU) x = fmaxf(x, 0.f); Cb[(long)r0 * ldc + ca0] = x;
    x = c01[v] * scale + bb1; if (RELU) x = fmaxf(x, 0.f); Cb[(long)r0 * ldc + ca1] = x;
    x = c10[v] * scale + bb0; if (RELU) x = fmaxf(x, 0.f); Cb[(long)r1 * ldc + ca0] = x;
    x = c11[v] * scale + bb1; if (RELU) x = fmaxf(x, 0.f); Cb[(long)r1 * ldc + ca1] = x;
  }
}

// ---------------------------------------------------------------------------
// Fused self-attention (softmax over keys). Q/K/V buffers laid out
// [token = l*B + b, 512] with head slice h*64. One workgroup handles one
// (b,h) and a 32-query block: scores (WMMA) -> LDS fp32 tile (32xS),
// row softmax in LDS, then O = P @ V with WMMA. S in {1024, 256}.
// ---------------------------------------------------------------------------
__global__ __launch_bounds__(256)
void attention_fused(const float* __restrict__ Qb, const float* __restrict__ Kb,
                     const float* __restrict__ Vb, const unsigned char* __restrict__ mask,
                     float* __restrict__ Ob, int S, int Bn, float scale) {
  extern __shared__ float smem[];
  float* sc  = smem;            // 32 * S floats
  float* red = smem + 32 * S;   // 32 * 8 floats
  int b = blockIdx.y >> 3;
  int h = blockIdx.y & 7;
  int q0 = blockIdx.x * 32;
  int lane = threadIdx.x & 31;
  int wave = threadIdx.x >> 5;
  long ldt = (long)Bn * D_MODEL;
  const float* Qp = Qb + (long)b * D_MODEL + h * DH;
  const float* Kp = Kb + (long)b * D_MODEL + h * DH;
  const float* Vp = Vb + (long)b * D_MODEL + h * DH;

  // phase 1: scores -> LDS
  for (int nc = wave * 32; nc < S; nc += 256) {
    v8f c00 = V8F_ZERO, c01 = V8F_ZERO, c10 = V8F_ZERO, c11 = V8F_ZERO;
#pragma unroll
    for (int k0 = 0; k0 < DH; k0 += 32) {
      v16h a0 = load_a_frag_g(Qp, ldt, q0, k0, lane);
      v16h a1 = load_a_frag_g(Qp, ldt, q0 + 16, k0, lane);
      v16h b0 = load_bt_frag_g(Kp, ldt, nc, k0, lane);
      v16h b1 = load_bt_frag_g(Kp, ldt, nc + 16, k0, lane);
      c00 = wmma16(a0, b0, c00);
      c01 = wmma16(a0, b1, c01);
      c10 = wmma16(a1, b0, c10);
      c11 = wmma16(a1, b1, c11);
    }
    int grp = lane >> 4, cl = lane & 15;
    int n0c = nc + cl, n1c = nc + 16 + cl;
    bool m0m = mask && mask[(long)b * S + n0c];
    bool m1m = mask && mask[(long)b * S + n1c];
#pragma unroll
    for (int v = 0; v < 8; ++v) {
      int r0 = grp * 8 + v, r1 = r0 + 16;
      float x;
      x = c00[v] * scale; if (m0m) x = -__builtin_inff(); sc[(long)r0 * S + n0c] = x;
      x = c01[v] * scale; if (m1m) x = -__builtin_inff(); sc[(long)r0 * S + n1c] = x;
      x = c10[v] * scale; if (m0m) x = -__builtin_inff(); sc[(long)r1 * S + n0c] = x;
      x = c11[v] * scale; if (m1m) x = -__builtin_inff(); sc[(long)r1 * S + n1c] = x;
    }
  }
  __syncthreads();

  // phase 2: softmax per row (8 threads per row)
  {
    int row = threadIdx.x >> 3;
    int sub = threadIdx.x & 7;
    float* rp = sc + (long)row * S;
    float mx = -__builtin_inff();
    for (int n = sub; n < S; n += 8) mx = fmaxf(mx, rp[n]);
    red[row * 8 + sub] = mx;
    __syncthreads();
    if (sub == 0) {
      float m2 = red[row * 8];
#pragma unroll
      for (int i = 1; i < 8; ++i) m2 = fmaxf(m2, red[row * 8 + i]);
      red[row * 8] = m2;
    }
    __syncthreads();
    mx = red[row * 8];
    __syncthreads();
    bool allneg = (mx == -__builtin_inff());
    float sum = 0.f;
    for (int n = sub; n < S; n += 8) {
      float e = allneg ? 0.f : __expf(rp[n] - mx);
      rp[n] = e;
      sum += e;
    }
    red[row * 8 + sub] = sum;
    __syncthreads();
    if (sub == 0) {
      float s2 = 0.f;
#pragma unroll
      for (int i = 0; i < 8; ++i) s2 += red[row * 8 + i];
      red[row * 8] = s2;
    }
    __syncthreads();
    float tot = red[row * 8];
    float inv = tot > 0.f ? 1.f / tot : 0.f;
    for (int n = sub; n < S; n += 8) rp[n] *= inv;
  }
  __syncthreads();

  // phase 3: O = P @ V ; 8 waves cover the 32x64 output (2x4 tiles of 16x16)
  {
    int mi = wave >> 2;   // 0..1
    int ni = wave & 3;    // 0..3
    v8f acc = V8F_ZERO;
    for (int kk = 0; kk < S; kk += 32) {
      v16h a = load_a_frag_lds(sc, S, mi * 16, kk, lane);
      v16h bv = load_bn_frag_g(Vp, ldt, ni * 16, kk, lane);
      acc = wmma16(a, bv, acc);
    }
    int grp = lane >> 4, cl = lane & 15;
#pragma unroll
    for (int v = 0; v < 8; ++v) {
      int qrow = q0 + mi * 16 + grp * 8 + v;
      int col = h * DH + ni * 16 + cl;
      Ob[((long)qrow * Bn + b) * D_MODEL + col] = acc[v];
    }
  }
}

// ---------------------------------------------------------------------------
// LayerNorm over D=512: out = (x1+x2 - mu)/sqrt(var+eps)*w + b, one row/block.
// ---------------------------------------------------------------------------
__global__ __launch_bounds__(256)
void layernorm_k(const float* __restrict__ x1, const float* __restrict__ x2,
                 const float* __restrict__ w, const float* __restrict__ bb,
                 float* __restrict__ out) {
  __shared__ float rbuf[256];
  long base = (long)blockIdx.x * D_MODEL;
  int t = threadIdx.x;
  float v0 = x1[base + t]       + (x2 ? x2[base + t]       : 0.f);
  float v1 = x1[base + t + 256] + (x2 ? x2[base + t + 256] : 0.f);
  rbuf[t] = v0 + v1;
  __syncthreads();
  for (int off = 128; off > 0; off >>= 1) {
    if (t < off) rbuf[t] += rbuf[t + off];
    __syncthreads();
  }
  float mu = rbuf[0] * (1.f / D_MODEL);
  __syncthreads();
  float d0 = v0 - mu, d1 = v1 - mu;
  rbuf[t] = d0 * d0 + d1 * d1;
  __syncthreads();
  for (int off = 128; off > 0; off >>= 1) {
    if (t < off) rbuf[t] += rbuf[t + off];
    __syncthreads();
  }
  float inv = rsqrtf(rbuf[0] * (1.f / D_MODEL) + 1e-5f);
  out[base + t]       = d0 * inv * w[t]       + bb[t];
  out[base + t + 256] = d1 * inv * w[t + 256] + bb[t + 256];
}

// ---------------------------------------------------------------------------
// Small elementwise kernels.
// ---------------------------------------------------------------------------
__global__ void add_k(const float* __restrict__ a, const float* __restrict__ b,
                      float* __restrict__ c, long n) {
  long i = (long)blockIdx.x * blockDim.x + threadIdx.x;
  if (i < n) c[i] = a[i] + b[i];
}

__global__ void add_qpos_k(const float* __restrict__ a, const float* __restrict__ qe,
                           float* __restrict__ c) {
  long i = (long)blockIdx.x * blockDim.x + threadIdx.x;
  long total = (long)NQUERY * BATCHB * D_MODEL;
  if (i >= total) return;
  int d = (int)(i & (D_MODEL - 1));
  int q = (int)(i / ((long)BATCHB * D_MODEL));
  float base = a ? a[i] : 0.f;
  c[i] = base + qe[(long)q * D_MODEL + d];
}

__global__ void fill0_k(float* __restrict__ p, long n) {
  long i = (long)blockIdx.x * blockDim.x + threadIdx.x;
  if (i < n) p[i] = 0.f;
}

// softmax over the q axis (assignment attention); one thread per (bh,n) column
__global__ void softmax_q_k(float* __restrict__ sc, const unsigned char* __restrict__ mask,
                            int Qn, int S) {
  long idx = (long)blockIdx.x * blockDim.x + threadIdx.x;
  long total = (long)BATCHB * NHEADS * S;
  if (idx >= total) return;
  int bh = (int)(idx / S);
  int n = (int)(idx % S);
  int b = bh >> 3;
  float* col = sc + (long)bh * Qn * S + n;
  if (mask && mask[(long)b * S + n]) {
    for (int q = 0; q < Qn; ++q) col[(long)q * S] = 0.f;
    return;
  }
  float mx = -__builtin_inff();
  for (int q = 0; q < Qn; ++q) mx = fmaxf(mx, col[(long)q * S]);
  float sum = 0.f;
  for (int q = 0; q < Qn; ++q) {
    float e = __expf(col[(long)q * S] - mx);
    col[(long)q * S] = e;
    sum += e;
  }
  float inv = sum > 0.f ? 1.f / sum : 0.f;
  for (int q = 0; q < Qn; ++q) col[(long)q * S] *= inv;
}

// mean over heads: out[b,q,n] = (1/8) sum_h sc[(b*8+h),q,n]; masked -> 0
__global__ void head_mean_k(const float* __restrict__ sc, const unsigned char* __restrict__ mask,
                            float* __restrict__ o, int Qn, int S) {
  long idx = (long)blockIdx.x * blockDim.x + threadIdx.x;
  long total = (long)BATCHB * Qn * S;
  if (idx >= total) return;
  int b = (int)(idx / ((long)Qn * S));
  long rem = idx % ((long)Qn * S);
  int q = (int)(rem / S);
  int n = (int)(rem % S);
  if (mask && mask[(long)b * S + n]) { o[idx] = 0.f; return; }
  float s = 0.f;
#pragma unroll
  for (int h = 0; h < NHEADS; ++h)
    s += sc[((long)(b * NHEADS + h) * Qn + q) * S + n];
  o[idx] = s * 0.125f;
}

// ---------------------------------------------------------------------------
// Host-side launch helper
// ---------------------------------------------------------------------------
static void launch_gemm(hipStream_t st, const float* A, long lda, long aB,
                        const float* B, long ldb, long bB,
                        const float* bias, float* C, long ldc, long cB,
                        int M, int N, int K, float scale, bool relu, bool bt, int batch) {
  dim3 g((unsigned)(N / 64), (unsigned)(M / 64), (unsigned)batch);
  dim3 blk(128);
  if (bt) {
    if (relu)
      gemm_wmma<true, true><<<g, blk, 0, st>>>(A, lda, aB, B, ldb, bB, bias, C, ldc, cB, K, scale);
    else
      gemm_wmma<true, false><<<g, blk, 0, st>>>(A, lda, aB, B, ldb, bB, bias, C, ldc, cB, K, scale);
  } else {
    gemm_wmma<false, false><<<g, blk, 0, st>>>(A, lda, aB, B, ldb, bB, bias, C, ldc, cB, K, scale);
  }
}

extern "C" void kernel_launch(void* const* d_in, const int* in_sizes, int n_in,
                              void* d_out_v, int out_size, void* d_ws, size_t ws_size,
                              hipStream_t stream) {
  (void)in_sizes; (void)n_in; (void)out_size; (void)ws_size;
  const float* src  = (const float*)d_in[0];
  const unsigned char* mask = (const unsigned char*)d_in[1];
  const float* qemb = (const float*)d_in[2];
  const float* pos  = (const float*)d_in[3];
  const float* enc_in_w  = (const float*)d_in[4];
  const float* enc_in_b  = (const float*)d_in[5];
  const float* enc_out_w = (const float*)d_in[6];
  const float* enc_out_b = (const float*)d_in[7];
  const float* enc_f1w = (const float*)d_in[8];
  const float* enc_f1b = (const float*)d_in[9];
  const float* enc_f2w = (const float*)d_in[10];
  const float* enc_f2b = (const float*)d_in[11];
  const float* enc_lnw = (const float*)d_in[12];
  const float* enc_lnb = (const float*)d_in[13];
  const float* dec_in_w  = (const float*)d_in[14];
  const float* dec_in_b  = (const float*)d_in[15];
  const float* dec_out_w = (const float*)d_in[16];
  const float* dec_out_b = (const float*)d_in[17];
  const float* dec_ca_w  = (const float*)d_in[18];
  const float* dec_ca_b  = (const float*)d_in[19];
  const float* dec_f1w = (const float*)d_in[20];
  const float* dec_f1b = (const float*)d_in[21];
  const float* dec_f2w = (const float*)d_in[22];
  const float* dec_f2b = (const float*)d_in[23];
  const float* dec_lnw = (const float*)d_in[24];
  const float* dec_lnb = (const float*)d_in[25];
  const float* fin_lnw = (const float*)d_in[26];
  const float* fin_lnb = (const float*)d_in[27];
  float* out = (float*)d_out_v;

  // workspace arena (floats)
  const long SZ_BIG = 8192L * 512;      // 4,194,304
  const long SZ_E   = 8192L * 2048;     // 16,777,216 (enc ffn hid / dec scores / dec ffn hid)
  const long SZ_DEC = 2048L * 512;      // 1,048,576
  float* ws  = (float*)d_ws;
  float* X   = ws;            // encoder stream; becomes `memory`
  float* QK  = X + SZ_BIG;    // out+pos (enc) ; memory+pos (dec, persistent)
  float* Qb  = QK + SZ_BIG;
  float* Kb  = Qb + SZ_BIG;
  float* Vb  = Kb + SZ_BIG;
  float* ATT = Vb + SZ_BIG;
  float* E   = ATT + SZ_BIG;
  float* TGT  = E + SZ_E;
  float* TGT2 = TGT + SZ_DEC;
  float* SA   = TGT2 + SZ_DEC;
  float* CB   = SA + SZ_DEC;
  float* DQK  = CB + SZ_DEC;

  (void)hipFuncSetAttribute((const void*)attention_fused,
                            hipFuncAttributeMaxDynamicSharedMemorySize,
                            (32 * SEQLEN + 256) * 4);

  const int TPB = 256;
  const long W2 = 512L * 512;   // 262144

  // =================== encoder ===================
  const float* cur = src;
  for (int i = 0; i < 6; ++i) {
    const float* wqkv = enc_in_w + (long)i * 3 * W2;
    const float* bqkv = enc_in_b + (long)i * 3 * 512;
    add_k<<<(unsigned)(SZ_BIG / TPB), TPB, 0, stream>>>(cur, pos, QK, SZ_BIG);
    launch_gemm(stream, QK, 512, 0, wqkv,            512, 0, bqkv,        Qb, 512, 0, 8192, 512, 512, 1.f, false, true, 1);
    launch_gemm(stream, QK, 512, 0, wqkv + W2,       512, 0, bqkv + 512,  Kb, 512, 0, 8192, 512, 512, 1.f, false, true, 1);
    launch_gemm(stream, cur, 512, 0, wqkv + 2 * W2,  512, 0, bqkv + 1024, Vb, 512, 0, 8192, 512, 512, 1.f, false, true, 1);
    attention_fused<<<dim3(SEQLEN / 32, 64), 256, (32 * (size_t)SEQLEN + 256) * 4, stream>>>(
        Qb, Kb, Vb, mask, ATT, SEQLEN, BATCHB, 0.125f);
    launch_gemm(stream, ATT, 512, 0, enc_out_w + (long)i * W2, 512, 0, enc_out_b + (long)i * 512,
                Qb, 512, 0, 8192, 512, 512, 1.f, false, true, 1);
    layernorm_k<<<8192, 256, 0, stream>>>(cur, Qb, enc_lnw + (long)(i * 2) * 512,
                                          enc_lnb + (long)(i * 2) * 512, X);
    launch_gemm(stream, X, 512, 0, enc_f1w + (long)i * DFF_ * 512, 512, 0, enc_f1b + (long)i * DFF_,
                E, DFF_, 0, 8192, DFF_, 512, 1.f, true, true, 1);
    launch_gemm(stream, E, DFF_, 0, enc_f2w + (long)i * 512 * DFF_, DFF_, 0, enc_f2b + (long)i * 512,
                Qb, 512, 0, 8192, 512, DFF_, 1.f, false, true, 1);
    layernorm_k<<<8192, 256, 0, stream>>>(X, Qb, enc_lnw + (long)(i * 2 + 1) * 512,
                                          enc_lnb + (long)(i * 2 + 1) * 512, X);
    cur = X;
  }
  // X now holds `memory` and must not be written below.

  // =================== decoder ===================
  fill0_k<<<(unsigned)(SZ_DEC / TPB), TPB, 0, stream>>>(TGT, SZ_DEC);
  add_k<<<(unsigned)(SZ_BIG / TPB), TPB, 0, stream>>>(X, pos, QK, SZ_BIG);  // memory + pos

  const long OUT0 = 1048576;     // out (Q*B*D)
  const long OUTP = OUT0;        // P_mean at offset 1,048,576
  const long OUTS = OUT0 + 2097152;  // S_mean after P_mean

  for (int i = 0; i < 6; ++i) {
    const float* wqkv = dec_in_w + (long)i * 3 * W2;
    const float* bqkv = dec_in_b + (long)i * 3 * 512;
    // self attention (no mask)
    add_qpos_k<<<(unsigned)(SZ_DEC / TPB), TPB, 0, stream>>>(TGT, qemb, DQK);
    launch_gemm(stream, DQK, 512, 0, wqkv,           512, 0, bqkv,        Qb, 512, 0, 2048, 512, 512, 1.f, false, true, 1);
    launch_gemm(stream, DQK, 512, 0, wqkv + W2,      512, 0, bqkv + 512,  Kb, 512, 0, 2048, 512, 512, 1.f, false, true, 1);
    launch_gemm(stream, TGT, 512, 0, wqkv + 2 * W2,  512, 0, bqkv + 1024, Vb, 512, 0, 2048, 512, 512, 1.f, false, true, 1);
    attention_fused<<<dim3(NQUERY / 32, 64), 256, (32 * (size_t)NQUERY + 256) * 4, stream>>>(
        Qb, Kb, Vb, nullptr, ATT, NQUERY, BATCHB, 0.125f);
    launch_gemm(stream, ATT, 512, 0, dec_out_w + (long)i * W2, 512, 0, dec_out_b + (long)i * 512,
                SA, 512, 0, 2048, 512, 512, 1.f, false, true, 1);
    layernorm_k<<<2048, 256, 0, stream>>>(TGT, SA, dec_lnw + (long)(i * 3) * 512,
                                          dec_lnb + (long)(i * 3) * 512, TGT2);
    // cross "assignment" attention
    const float* caw = dec_ca_w + (long)i * 4 * W2;
    const float* cab = dec_ca_b + (long)i * 4 * 512;
    add_qpos_k<<<(unsigned)(SZ_DEC / TPB), TPB, 0, stream>>>(SA, qemb, DQK);
    launch_gemm(stream, DQK, 512, 0, caw,           512, 0, cab,        Qb, 512, 0, 2048, 512, 512, 1.f, false, true, 1);
    launch_gemm(stream, QK,  512, 0, caw + W2,      512, 0, cab + 512,  Kb, 512, 0, 8192, 512, 512, 1.f, false, true, 1);
    launch_gemm(stream, X,   512, 0, caw + 2 * W2,  512, 0, cab + 1024, Vb, 512, 0, 8192, 512, 512, 1.f, false, true, 1);
    // scores[bh, q, n] = q . k / 8 : batched over bh=64 (per-head offset is 64*bh)
    launch_gemm(stream, Qb, 4096, 64, Kb, 4096, 64, nullptr,
                E, 1024, 262144, NQUERY, SEQLEN, 64, 0.125f, false, true, 64);
    if (i == 5)
      head_mean_k<<<8192, 256, 0, stream>>>(E, mask, out + OUTS, NQUERY, SEQLEN);
    softmax_q_k<<<(unsigned)((64L * SEQLEN) / TPB), TPB, 0, stream>>>(E, mask, NQUERY, SEQLEN);
    if (i == 5)
      head_mean_k<<<8192, 256, 0, stream>>>(E, mask, out + OUTP, NQUERY, SEQLEN);
    // ca_pre = P @ V (B non-transposed), batched over bh
    launch_gemm(stream, E, 1024, 262144, Vb, 4096, 64, nullptr,
                ATT, 4096, 64, NQUERY, 64, SEQLEN, 1.f, false, false, 64);
    launch_gemm(stream, ATT, 512, 0, caw + 3 * W2, 512, 0, cab + 1536,
                CB, 512, 0, 2048, 512, 512, 1.f, false, true, 1);
    layernorm_k<<<2048, 256, 0, stream>>>(TGT2, CB, dec_lnw + (long)(i * 3 + 1) * 512,
                                          dec_lnb + (long)(i * 3 + 1) * 512, TGT);
    // ffn
    launch_gemm(stream, TGT, 512, 0, dec_f1w + (long)i * DFF_ * 512, 512, 0, dec_f1b + (long)i * DFF_,
                E, DFF_, 0, 2048, DFF_, 512, 1.f, true, true, 1);
    launch_gemm(stream, E, DFF_, 0, dec_f2w + (long)i * 512 * DFF_, DFF_, 0, dec_f2b + (long)i * 512,
                SA, 512, 0, 2048, 512, DFF_, 1.f, false, true, 1);
    layernorm_k<<<2048, 256, 0, stream>>>(TGT, SA, dec_lnw + (long)(i * 3 + 2) * 512,
                                          dec_lnb + (long)(i * 3 + 2) * 512, TGT);
  }
  // final layer norm -> out[0 .. Q*B*D)
  layernorm_k<<<2048, 256, 0, stream>>>(TGT, nullptr, fin_lnw, fin_lnb, out);
}
